// Graph_encoder_sample_weight_23003844837988
// MI455X (gfx1250) — matrix operand order repaired and checked by reading, not compiled, not run on hardware
//
#include <hip/hip_runtime.h>
#include <hip/hip_bf16.h>

// ---------------- constants (match reference) ----------------
#define NN   32768      // B*N nodes
#define DD   256        // hidden dim
#define D4   1024       // 4*D (LSTM gates)
#define EE   524288     // edges
#define BB   64         // batch
#define SEQ  512        // nodes per graph
#define MCH  8192       // LSTM row chunk
#define KDIM 256        // all GEMMs have K = 256

typedef __attribute__((ext_vector_type(16))) __bf16 v16bf;
typedef __attribute__((ext_vector_type(8)))  __bf16 v8bf;
typedef __attribute__((ext_vector_type(8)))  float  v8f;

__device__ __forceinline__ unsigned short f2bf(float f) {
    union { float f; unsigned int u; } x; x.f = f;
    unsigned int u = x.u;
    u += 0x7FFFu + ((u >> 16) & 1u);   // round-to-nearest-even
    return (unsigned short)(u >> 16);
}

__device__ __forceinline__ float sigm(float x) { return 1.0f / (1.0f + expf(-x)); }

// ---------------- WMMA fragment loads (contiguous k = 16*(L>>4)+s, same in A and B) ----------------
// A (16x32 bf16, row-major, ld=KDIM): lane L<16 -> row L, K = k0+[0..15]
//                                     lane L>=16 -> row L-16, K = k0+16+[0..15]
__device__ __forceinline__ v16bf load_a(const __bf16* __restrict__ A, int row, int k0, int lane) {
    const int r    = row + (lane & 15);
    const int koff = (lane >> 4) * 16;
    const __bf16* p = A + (size_t)r * KDIM + k0 + koff;
    v8bf lo = *(const v8bf*)p;
    v8bf hi = *(const v8bf*)(p + 8);
    v16bf v;
#pragma unroll
    for (int i = 0; i < 8; ++i) { v[i] = lo[i]; v[i + 8] = hi[i]; }
    return v;
}

// B stored transposed as Bt[N][K] row-major: lane holds column n = base+(L&15),
// 16 consecutive K values starting at k0 + 16*(L>>4)  -> one 32B load.
__device__ __forceinline__ v16bf load_b(const __bf16* __restrict__ Bt, int col, int k0, int lane) {
    const int n  = col + (lane & 15);
    const int kb = k0 + (lane >> 4) * 16;
    return *(const v16bf*)(Bt + (size_t)n * KDIM + kb);
}

// ---------------- dual-input WMMA GEMM ----------------
// C[M,N] = A1[M,K]*B1t[N,K]^T (+ A2*B2t^T) + bias, optional tanh.
// Writes f32 (Cf) and/or bf16 (Cb). Block: 256 thr = 8 waves (2M x 4N of 32x32 tiles).
__global__ __launch_bounds__(256)
void k_gemm(const __bf16* __restrict__ A1, const __bf16* __restrict__ B1,
            const __bf16* __restrict__ A2, const __bf16* __restrict__ B2,
            const float* __restrict__ bias, float* __restrict__ Cf,
            unsigned short* __restrict__ Cb, int M, int N, int act)
{
    const int lane   = threadIdx.x & 31;
    const int wave   = threadIdx.x >> 5;
    const int m_base = blockIdx.x * 64  + (wave & 1) * 32;
    const int n_base = blockIdx.y * 128 + (wave >> 1) * 32;
    if (m_base >= M || n_base >= N) return;

    v8f acc[2][2] = {};

    for (int k0 = 0; k0 < KDIM; k0 += 32) {
        v16bf a0 = load_a(A1, m_base,      k0, lane);
        v16bf a1 = load_a(A1, m_base + 16, k0, lane);
        v16bf b0 = load_b(B1, n_base,      k0, lane);
        v16bf b1 = load_b(B1, n_base + 16, k0, lane);
        acc[0][0] = __builtin_amdgcn_wmma_f32_16x16x32_bf16(false, a0, false, b0, (short)0, acc[0][0], false, false);
        acc[0][1] = __builtin_amdgcn_wmma_f32_16x16x32_bf16(false, a0, false, b1, (short)0, acc[0][1], false, false);
        acc[1][0] = __builtin_amdgcn_wmma_f32_16x16x32_bf16(false, a1, false, b0, (short)0, acc[1][0], false, false);
        acc[1][1] = __builtin_amdgcn_wmma_f32_16x16x32_bf16(false, a1, false, b1, (short)0, acc[1][1], false, false);
    }
    if (A2) {
        for (int k0 = 0; k0 < KDIM; k0 += 32) {
            v16bf a0 = load_a(A2, m_base,      k0, lane);
            v16bf a1 = load_a(A2, m_base + 16, k0, lane);
            v16bf b0 = load_b(B2, n_base,      k0, lane);
            v16bf b1 = load_b(B2, n_base + 16, k0, lane);
            acc[0][0] = __builtin_amdgcn_wmma_f32_16x16x32_bf16(false, a0, false, b0, (short)0, acc[0][0], false, false);
            acc[0][1] = __builtin_amdgcn_wmma_f32_16x16x32_bf16(false, a0, false, b1, (short)0, acc[0][1], false, false);
            acc[1][0] = __builtin_amdgcn_wmma_f32_16x16x32_bf16(false, a1, false, b0, (short)0, acc[1][0], false, false);
            acc[1][1] = __builtin_amdgcn_wmma_f32_16x16x32_bf16(false, a1, false, b1, (short)0, acc[1][1], false, false);
        }
    }

    // C/D layout: n = lane&15, m = vgpr_idx + 8*(lane>>4)
    const int nloc = lane & 15;
    const int roff = (lane >> 4) * 8;
#pragma unroll
    for (int i = 0; i < 2; ++i) {
#pragma unroll
        for (int j = 0; j < 2; ++j) {
            const int n = n_base + j * 16 + nloc;
            const float bv = bias ? bias[n] : 0.0f;
#pragma unroll
            for (int r = 0; r < 8; ++r) {
                const int m = m_base + i * 16 + roff + r;
                float v = acc[i][j][r] + bv;
                if (act) v = tanhf(v);
                const size_t idx = (size_t)m * N + n;
                if (Cf) Cf[idx] = v;
                if (Cb) Cb[idx] = f2bf(v);
            }
        }
    }
}

// ---------------- small helper kernels ----------------
__global__ __launch_bounds__(256) void k_fill(float* p, float v) {
    p[(size_t)blockIdx.x * 256 + threadIdx.x] = v;
}

// transpose+convert DxD f32 [K,N] -> bf16 [N,K]
__global__ __launch_bounds__(256) void k_convT(const float* __restrict__ in, unsigned short* __restrict__ out) {
    const int t = blockIdx.x * 256 + threadIdx.x;   // t = n*256 + k
    const int n = t >> 8, k = t & 255;
    out[t] = f2bf(in[(k << 8) + n]);
}

__global__ __launch_bounds__(256) void k_conv(const float* __restrict__ in, unsigned short* __restrict__ out) {
    const size_t t = (size_t)blockIdx.x * 256 + threadIdx.x;
    out[t] = f2bf(in[t]);
}

__global__ __launch_bounds__(256) void k_bias2(const float* a, const float* b, float* out) {
    const int t = blockIdx.x * 256 + threadIdx.x;
    out[t] = a[t] + b[t];
}

__global__ __launch_bounds__(256)
void k_embed(const int* __restrict__ tok, const int* __restrict__ wty,
             const float* __restrict__ emb, const float* __restrict__ wtt,
             unsigned short* __restrict__ out)
{
    const size_t t = (size_t)blockIdx.x * 256 + threadIdx.x;   // NN*DD threads
    const int node = (int)(t >> 8), c = (int)(t & 255);
    const float v = emb[(size_t)tok[node] * DD + c] + wtt[(size_t)wty[node] * DD + c];
    out[t] = f2bf(v);
}

__global__ __launch_bounds__(256)
void k_deg(const int* __restrict__ dst, const float* __restrict__ ew, float* __restrict__ deg) {
    const int e = blockIdx.x * 256 + threadIdx.x;
    atomicAdd(&deg[dst[e]], ew[e]);
}

__global__ __launch_bounds__(256) void k_rsqrt(float* p) {
    const int t = blockIdx.x * 256 + threadIdx.x;
    p[t] = rsqrtf(p[t]);
}

// 16 threads/edge, 16 cols each: ACC[dst] += dinv[s]*ew*dinv[d] * XW[src]
__global__ __launch_bounds__(256)
void k_edge_scatter(const int* __restrict__ src, const int* __restrict__ dst,
                    const float* __restrict__ ew, const float* __restrict__ dinv,
                    const float* __restrict__ XW, float* __restrict__ ACC)
{
    const int t  = blockIdx.x * 256 + threadIdx.x;
    const int e  = t >> 4;
    const int c0 = (t & 15) * 16;
    const int s = src[e], d = dst[e];
    const float w = dinv[s] * ew[e] * dinv[d];
    const float4* xs = (const float4*)(XW + (size_t)s * DD + c0);
    float* ad = ACC + (size_t)d * DD + c0;
#pragma unroll
    for (int q = 0; q < 4; ++q) {
        const float4 v = xs[q];
        atomicAdd(ad + q * 4 + 0, w * v.x);
        atomicAdd(ad + q * 4 + 1, w * v.y);
        atomicAdd(ad + q * 4 + 2, w * v.z);
        atomicAdd(ad + q * 4 + 3, w * v.w);
    }
}

__global__ __launch_bounds__(256)
void k_gcn_combine(const float* __restrict__ ACC, const float* __restrict__ XW,
                   const float* __restrict__ dinv, const float* __restrict__ bias,
                   unsigned short* __restrict__ outb)
{
    const size_t t = (size_t)blockIdx.x * 256 + threadIdx.x;
    const int node = (int)(t >> 8), c = (int)(t & 255);
    const float di = dinv[node];
    outb[t] = f2bf(ACC[t] + di * di * XW[t] + bias[c]);
}

// gates layout [rows,4D] as i|f|g|o blocks of D (torch order)
__global__ __launch_bounds__(256)
void k_lstm_elem(const float* __restrict__ g, const float* __restrict__ cprev,
                 float* __restrict__ cout, float* __restrict__ hf, unsigned short* __restrict__ hb)
{
    const size_t t = (size_t)blockIdx.x * 256 + threadIdx.x;   // rows*DD
    const size_t m = t >> 8; const int c = (int)(t & 255);
    const float* gr = g + (m << 10);
    const float gi = gr[c], gf_ = gr[256 + c], gg = gr[512 + c], go = gr[768 + c];
    const float cp = cprev ? cprev[t] : 0.0f;
    const float cn = sigm(gf_) * cp + sigm(gi) * tanhf(gg);
    const float h  = sigm(go) * tanhf(cn);
    cout[t] = cn;
    if (hf) hf[t] = h;
    if (hb) hb[t] = f2bf(h);
}

__global__ __launch_bounds__(256)
void k_score(const float* __restrict__ HT, const float* __restrict__ Wp2,
             const float* __restrict__ ms, float* __restrict__ sc)
{
    const int n = blockIdx.x * 256 + threadIdx.x;   // NN threads
    const float* h = HT + (size_t)n * DD;
    float s = 0.0f;
#pragma unroll 8
    for (int c = 0; c < DD; ++c) s += h[c] * Wp2[c];
    sc[n] = s + ms[n];
}

// one block per batch: softmax over SEQ scores + weighted column sum
__global__ __launch_bounds__(256)
void k_pool(const float* __restrict__ score, const float* __restrict__ X,
            float* __restrict__ gf, unsigned short* __restrict__ gb)
{
    __shared__ float sh[SEQ];
    __shared__ float red[256];
    const int b = blockIdx.x, t = threadIdx.x;
    const float s0 = score[b * SEQ + t];
    const float s1 = score[b * SEQ + 256 + t];
    red[t] = fmaxf(s0, s1);
    __syncthreads();
    for (int s = 128; s > 0; s >>= 1) { if (t < s) red[t] = fmaxf(red[t], red[t + s]); __syncthreads(); }
    const float mx = red[0];
    __syncthreads();
    const float e0 = expf(s0 - mx), e1 = expf(s1 - mx);
    sh[t] = e0; sh[256 + t] = e1;
    red[t] = e0 + e1;
    __syncthreads();
    for (int s = 128; s > 0; s >>= 1) { if (t < s) red[t] += red[t + s]; __syncthreads(); }
    const float inv = 1.0f / red[0];
    float acc = 0.0f;
    const float* Xp = X + ((size_t)b * SEQ) * DD + t;
    for (int n = 0; n < SEQ; ++n) acc += sh[n] * Xp[(size_t)n * DD];
    acc *= inv;
    gf[b * DD + t] = acc;
    gb[b * DD + t] = f2bf(acc);
}

__global__ __launch_bounds__(256)
void k_xout(const float* __restrict__ X2, const float* __restrict__ maskp, float* __restrict__ out)
{
    const size_t t = (size_t)blockIdx.x * 256 + threadIdx.x;
    out[t] = tanhf(X2[t]) * maskp[t >> 8];
}

// ---------------- host orchestration ----------------
extern "C" void kernel_launch(void* const* d_in, const int* in_sizes, int n_in,
                              void* d_out, int out_size, void* d_ws, size_t ws_size,
                              hipStream_t stream)
{
    const int*   tok   = (const int*)d_in[0];
    const int*   wty   = (const int*)d_in[1];
    const int*   src   = (const int*)d_in[2];
    const int*   dst   = src + EE;
    const float* ew    = (const float*)d_in[3];
    const float* maskp = (const float*)d_in[4];
    const float* masks = (const float*)d_in[5];
    const float* emb   = (const float*)d_in[6];
    const float* wtt   = (const float*)d_in[7];
    const float* W_w   = (const float*)d_in[8];   const float* b_w  = (const float*)d_in[9];
    const float* W_g1  = (const float*)d_in[10];  const float* b_g1 = (const float*)d_in[11];
    const float* W_g2  = (const float*)d_in[12];  const float* b_g2 = (const float*)d_in[13];
    const float* W_p1  = (const float*)d_in[14];  const float* b_p1 = (const float*)d_in[15];
    const float* W_p2  = (const float*)d_in[16];
    const float* Wih_x = (const float*)d_in[17];  const float* Whh_x = (const float*)d_in[18];
    const float* bih_x = (const float*)d_in[19];  const float* bhh_x = (const float*)d_in[20];
    const float* Wih_g = (const float*)d_in[21];  const float* Whh_g = (const float*)d_in[22];
    const float* bih_g = (const float*)d_in[23];  const float* bhh_g = (const float*)d_in[24];

    // workspace arena (256B-aligned slots)
    char* w = (char*)d_ws;
    size_t off = 0;
    auto alloc = [&](size_t bytes) -> char* {
        char* p = w + off; off = (off + bytes + 255) & ~(size_t)255; return p;
    };
    unsigned short* XB   = (unsigned short*)alloc((size_t)NN * DD * 2);   // x (bf16)
    unsigned short* MIDB = (unsigned short*)alloc((size_t)NN * DD * 2);   // embed tmp / gcn out (bf16)
    float*          XW   = (float*)alloc((size_t)NN * DD * 4);            // gcn xw / pool ht
    float*          ACC  = (float*)alloc((size_t)NN * DD * 4);            // gcn accum, aliased as X2F
    float*          GATES= (float*)alloc((size_t)MCH * D4 * 4);           // lstm gates (chunked) / graph gates
    float*          CST  = (float*)alloc((size_t)MCH * DD * 4);           // lstm cell state / graph c
    unsigned short* HB   = (unsigned short*)alloc((size_t)MCH * DD * 2);  // lstm h1 bf16 / graph h bf16
    float*          X1F  = (float*)alloc((size_t)NN * DD * 4);
    unsigned short* X1B  = (unsigned short*)alloc((size_t)NN * DD * 2);
    unsigned short* X2B  = (unsigned short*)alloc((size_t)NN * DD * 2);
    float*          DINV = (float*)alloc((size_t)NN * 4);
    float*          SCORE= (float*)alloc((size_t)NN * 4);
    unsigned short* WWT  = (unsigned short*)alloc((size_t)DD * DD * 2);
    unsigned short* WG1T = (unsigned short*)alloc((size_t)DD * DD * 2);
    unsigned short* WG2T = (unsigned short*)alloc((size_t)DD * DD * 2);
    unsigned short* WP1T = (unsigned short*)alloc((size_t)DD * DD * 2);
    unsigned short* WIHX = (unsigned short*)alloc((size_t)D4 * DD * 2);
    unsigned short* WHHX = (unsigned short*)alloc((size_t)D4 * DD * 2);
    unsigned short* WIHG = (unsigned short*)alloc((size_t)D4 * DD * 2);
    unsigned short* WHHG = (unsigned short*)alloc((size_t)D4 * DD * 2);
    float*          BIASX= (float*)alloc((size_t)D4 * 4);
    float*          BIASG= (float*)alloc((size_t)D4 * 4);
    float*          G1F  = (float*)alloc((size_t)BB * DD * 4);
    unsigned short* G1B  = (unsigned short*)alloc((size_t)BB * DD * 2);
    float*          G2F  = (float*)alloc((size_t)BB * DD * 4);
    unsigned short* G2B  = (unsigned short*)alloc((size_t)BB * DD * 2);
    float* X2F = ACC;                       // alias: ACC is dead after GCN2 combine
    if (off > ws_size) return;              // workspace too small: bail

    auto gemm = [&](const unsigned short* A1, const unsigned short* B1,
                    const unsigned short* A2, const unsigned short* B2,
                    const float* bias, float* Cf, unsigned short* Cb,
                    int M, int N, int act) {
        dim3 g(M / 64, N / 128);
        k_gemm<<<g, 256, 0, stream>>>((const __bf16*)A1, (const __bf16*)B1,
                                      (const __bf16*)A2, (const __bf16*)B2,
                                      bias, Cf, Cb, M, N, act);
    };

    // --- weight prep (bf16 / transposed) ---
    k_convT<<<256, 256, 0, stream>>>(W_w,  WWT);
    k_convT<<<256, 256, 0, stream>>>(W_g1, WG1T);
    k_convT<<<256, 256, 0, stream>>>(W_g2, WG2T);
    k_convT<<<256, 256, 0, stream>>>(W_p1, WP1T);
    k_conv <<<1024, 256, 0, stream>>>(Wih_x, WIHX);
    k_conv <<<1024, 256, 0, stream>>>(Whh_x, WHHX);
    k_conv <<<1024, 256, 0, stream>>>(Wih_g, WIHG);
    k_conv <<<1024, 256, 0, stream>>>(Whh_g, WHHG);
    k_bias2<<<4, 256, 0, stream>>>(bih_x, bhh_x, BIASX);
    k_bias2<<<4, 256, 0, stream>>>(bih_g, bhh_g, BIASG);

    // --- degree normalization (shared by both GCN layers) ---
    k_fill <<<NN / 256, 256, 0, stream>>>(DINV, 1.0f);
    k_deg  <<<EE / 256, 256, 0, stream>>>(dst, ew, DINV);
    k_rsqrt<<<NN / 256, 256, 0, stream>>>(DINV);

    // --- embedding + input projection: XB = bf16(embed @ W_w + b_w) ---
    k_embed<<<NN, 256, 0, stream>>>(tok, wty, emb, wtt, MIDB);
    gemm(MIDB, WWT, nullptr, nullptr, b_w, nullptr, XB, NN, DD, 0);

    auto gcn = [&](const unsigned short* Xin, const unsigned short* Wt,
                   const float* bg, unsigned short* Mout) {
        gemm(Xin, Wt, nullptr, nullptr, nullptr, XW, nullptr, NN, DD, 0);
        k_fill<<<NN, 256, 0, stream>>>(ACC, 0.0f);                  // NN*DD elements
        k_edge_scatter<<<(EE * 16) / 256, 256, 0, stream>>>(src, dst, ew, DINV, XW, ACC);
        k_gcn_combine<<<NN, 256, 0, stream>>>(ACC, XW, DINV, bg, Mout);
    };

    auto lstm = [&](const unsigned short* Xin, const unsigned short* Min,
                    float* Hf, unsigned short* Hbo) {
        for (int ch = 0; ch < NN / MCH; ++ch) {
            const size_t o = (size_t)ch * MCH * DD;
            gemm(Xin + o, WIHX, nullptr, nullptr, BIASX, GATES, nullptr, MCH, D4, 0);
            k_lstm_elem<<<MCH, 256, 0, stream>>>(GATES, nullptr, CST, nullptr, HB);
            gemm(Min + o, WIHX, HB, WHHX, BIASX, GATES, nullptr, MCH, D4, 0);
            k_lstm_elem<<<MCH, 256, 0, stream>>>(GATES, CST, CST, Hf + o, Hbo + o);
        }
    };

    auto pool = [&](const unsigned short* Xb, const float* Xf,
                    float* gf, unsigned short* gb) {
        gemm(Xb, WP1T, nullptr, nullptr, b_p1, XW, nullptr, NN, DD, 1);   // tanh
        k_score<<<NN / 256, 256, 0, stream>>>(XW, W_p2, masks, SCORE);
        k_pool<<<BB, 256, 0, stream>>>(SCORE, Xf, gf, gb);
    };

    // --- round 1 ---
    gcn(XB, WG1T, b_g1, MIDB);
    lstm(XB, MIDB, X1F, X1B);
    pool(X1B, X1F, G1F, G1B);

    // --- round 2 ---
    gcn(X1B, WG2T, b_g2, MIDB);
    lstm(X1B, MIDB, X2F, X2B);
    pool(X2B, X2F, G2F, G2B);

    // --- graph-level LSTM (M=64) ---
    float* gout = (float*)d_out + (size_t)NN * DD;
    gemm(G1B, WIHG, nullptr, nullptr, BIASG, GATES, nullptr, BB, D4, 0);
    k_lstm_elem<<<BB, 256, 0, stream>>>(GATES, nullptr, CST, nullptr, HB);
    gemm(G2B, WIHG, HB, WHHG, BIASG, GATES, nullptr, BB, D4, 0);
    k_lstm_elem<<<BB, 256, 0, stream>>>(GATES, CST, CST, gout, nullptr);

    // --- node output: tanh(x2) * mask_pad ---
    k_xout<<<NN, 256, 0, stream>>>(X2F, maskp, (float*)d_out);
}
